// Model_2662879724014
// MI455X (gfx1250) — compile-verified, matrix operand order checked by reading, not run on hardware
//
#include <hip/hip_runtime.h>
#include <hip/hip_bf16.h>
#include <math.h>

// ---------------------------------------------------------------------------
// nGPT transformer block for MI455X (gfx1250, wave32, WMMA).
// All matmuls: v_wmma_f32_16x16x32_bf16 (D = A x B + C, f32 accumulate).
// GEMM staging: global_load_async_to_lds_b128 (ASYNCcnt) + double buffering.
// Assumed operand layouts (CDNA5 ISA 7.12.2):
//   A (16x32 bf16): lane<16 -> row M=lane, K = {0..7, 16..23}; lane>=16 -> row
//                   M=lane-16, K = {8..15, 24..31}  => two contiguous 16B loads
//   B (32x16 bf16): lane n holds column N=n with the same K striping, so we
//                   store weights transposed (Bt[n][k]) and load like A.
//   C/D (16x16 f32): lane<16 -> N=lane, VGPR r -> M=r; lane>=16 -> M=8+r.
// ---------------------------------------------------------------------------

typedef __bf16 bf16;
typedef __attribute__((ext_vector_type(16))) __bf16 v16bf;
typedef __attribute__((ext_vector_type(8)))  __bf16 v8bf;
typedef __attribute__((ext_vector_type(8)))  float  v8f;

#define BATCH 2
#define SEQ   2048
#define DIM   1024
#define HEADS 16
#define HD    64
#define HID   2730
#define HIDP  2816          // HID padded to multiple of 128
#define NTOK  (BATCH*SEQ)   // 4096
#define SQRT_DIM 32.0f
#define SQRT_HD   8.0f

__device__ __forceinline__ v8f wmma_bf16(v16bf a, v16bf b, v8f c) {
  return __builtin_amdgcn_wmma_f32_16x16x32_bf16(false, a, false, b, (short)0, c,
                                                 false, false);
}

__device__ __forceinline__ v16bf load_frag(const bf16* p0, const bf16* p1) {
  v8bf lo = *(const v8bf*)p0;   // K k..k+7
  v8bf hi = *(const v8bf*)p1;   // K k+16..k+23
  v16bf r;
#pragma unroll
  for (int i = 0; i < 8; ++i) { r[i] = lo[i]; r[i + 8] = hi[i]; }
  return r;
}

__device__ __forceinline__ v8f zero8() {
  v8f z;
#pragma unroll
  for (int i = 0; i < 8; ++i) z[i] = 0.0f;
  return z;
}

// Per-lane async DMA of 16 bytes: global -> LDS, tracked by ASYNCcnt.
// lds generic pointer's low 32 bits are the wave-relative LDS byte address.
__device__ __forceinline__ void async_copy_b128(void* lds, const void* gaddr) {
  unsigned la = (unsigned)(size_t)lds;
  asm volatile("global_load_async_to_lds_b128 %0, %1, off"
               :: "v"(la), "v"(gaddr) : "memory");
}
__device__ __forceinline__ void wait_async0() {
  asm volatile("s_wait_asynccnt 0x0" ::: "memory");
}

// ---------------------------------------------------------------------------
// Generic bf16 WMMA GEMM: C[M,N] = A[M,K] * Bt[N,K]^T  (A,Bt row-major bf16).
// M % 128 == 0, N % 128 == 0, K % 64 == 0. 256 threads = 8 waves.
// Block tile 128x128; K in chunks of 64, double-buffered in LDS via async DMA.
// Wave tile 32x64 (2x4 WMMA accumulators).
// ---------------------------------------------------------------------------
#define GBM 128
#define GBN 128
#define GKC 64
#define LDS_S 72   // padded LDS row stride (bf16 elems); 144B, 16B aligned

__global__ __launch_bounds__(256) void gemm_bf16_wmma(
    const bf16* __restrict__ A, const bf16* __restrict__ Bt,
    float* __restrict__ C, int M, int N, int K) {
  __shared__ __attribute__((aligned(16))) bf16 As[2][GBM * LDS_S];
  __shared__ __attribute__((aligned(16))) bf16 Bs[2][GBN * LDS_S];

  const int tid  = threadIdx.x;
  const int wave = tid >> 5;
  const int lane = tid & 31;
  const int half = lane >> 4;
  const int lr   = lane & 15;
  const int m0 = blockIdx.y * GBM;
  const int n0 = blockIdx.x * GBN;
  const int wm = (wave >> 1) * 32;   // wave M offset within block tile
  const int wn = (wave & 1) * 64;    // wave N offset within block tile

  // issue async fills of one K-chunk (128x64 bf16 per matrix) into buffer buf
  auto stage = [&](int buf, int kk) {
#pragma unroll
    for (int i = 0; i < 4; ++i) {
      int chunk = tid + i * 256;          // 0..1023 chunks of 8 bf16
      int row   = chunk >> 3;             // 0..127
      int c8    = (chunk & 7) * 8;        // 0..56
      async_copy_b128(&As[buf][row * LDS_S + c8],
                      &A[(size_t)(m0 + row) * K + kk + c8]);
      async_copy_b128(&Bs[buf][row * LDS_S + c8],
                      &Bt[(size_t)(n0 + row) * K + kk + c8]);
    }
  };

  v8f acc[2][4];
#pragma unroll
  for (int i = 0; i < 2; ++i)
#pragma unroll
    for (int j = 0; j < 4; ++j) acc[i][j] = zero8();

  stage(0, 0);
  wait_async0();
  __syncthreads();

  int buf = 0;
  for (int kk = 0; kk < K; kk += GKC) {
    if (kk + GKC < K) stage(buf ^ 1, kk + GKC);   // overlap DMA with WMMA

#pragma unroll
    for (int ks = 0; ks < GKC; ks += 32) {
      v16bf af[2], bfm[4];
#pragma unroll
      for (int i = 0; i < 2; ++i) {
        const bf16* p = &As[buf][(wm + i * 16 + lr) * LDS_S + ks + half * 8];
        af[i] = load_frag(p, p + 16);
      }
#pragma unroll
      for (int j = 0; j < 4; ++j) {
        const bf16* p = &Bs[buf][(wn + j * 16 + lr) * LDS_S + ks + half * 8];
        bfm[j] = load_frag(p, p + 16);
      }
#pragma unroll
      for (int i = 0; i < 2; ++i)
#pragma unroll
        for (int j = 0; j < 4; ++j) acc[i][j] = wmma_bf16(af[i], bfm[j], acc[i][j]);
    }

    wait_async0();       // next tile resident
    __syncthreads();     // all waves done reading current tile
    buf ^= 1;
  }

#pragma unroll
  for (int i = 0; i < 2; ++i) {
    int rbase = m0 + wm + i * 16 + half * 8;
#pragma unroll
    for (int j = 0; j < 4; ++j) {
      int col = n0 + wn + j * 16 + lr;
#pragma unroll
      for (int r = 0; r < 8; ++r)
        C[(size_t)(rbase + r) * N + col] = acc[i][j][r];
    }
  }
}

// ---------------------------------------------------------------------------
// Elementwise / conversion kernels
// ---------------------------------------------------------------------------
__global__ void cvt_f32_bf16(const float* __restrict__ src, bf16* __restrict__ dst,
                             size_t n) {
  size_t i = (size_t)blockIdx.x * 256 + threadIdx.x;
  if (i < n) dst[i] = (bf16)src[i];
}

// W[K,N] f32 -> Wt[Np,Kp] bf16 (transposed, zero padded)
__global__ void transpose_pad_bf16(const float* __restrict__ W, bf16* __restrict__ Wt,
                                   int K, int N, int Kp, int Np) {
  size_t i = (size_t)blockIdx.x * 256 + threadIdx.x;
  size_t total = (size_t)Np * Kp;
  if (i >= total) return;
  int k = (int)(i % Kp);
  int n = (int)(i / Kp);
  float v = (n < N && k < K) ? W[(size_t)k * N + n] : 0.0f;
  Wt[i] = (bf16)v;
}

// V f32 [B,S,H*HD] -> Vt bf16 [B,H,HD,S]
__global__ void v_transpose_bf16(const float* __restrict__ V, bf16* __restrict__ Vt) {
  size_t i = (size_t)blockIdx.x * 256 + threadIdx.x;
  size_t total = (size_t)BATCH * HEADS * HD * SEQ;
  if (i >= total) return;
  int s = (int)(i % SEQ);
  size_t t = i / SEQ;
  int d = (int)(t % HD); t /= HD;
  int h = (int)(t % HEADS);
  int b = (int)(t / HEADS);
  Vt[i] = (bf16)V[((size_t)(b * SEQ + s)) * DIM + h * HD + d];
}

// ---------------------------------------------------------------------------
// RoPE + cosine-norm + eff_s_qk scale.  One wave per (token, head).
// X f32 [B,S,H*HD] -> Out bf16 [B,H,S,HD]
// ---------------------------------------------------------------------------
__global__ __launch_bounds__(256) void rope_cosnorm(
    const float* __restrict__ X, const float* __restrict__ s_qk,
    bf16* __restrict__ Out) {
  const int wave = threadIdx.x >> 5;
  const int lane = threadIdx.x & 31;
  size_t pair = (size_t)blockIdx.x * 8 + wave;      // over NTOK*HEADS
  if (pair >= (size_t)NTOK * HEADS) return;
  int h = (int)(pair % HEADS);
  size_t bs = pair / HEADS;                          // token index
  int b = (int)(bs / SEQ);
  int s = (int)(bs % SEQ);

  const float* xp = X + bs * DIM + h * HD;
  float x0 = xp[lane];
  float x1 = xp[lane + 32];
  // inv_freq[lane] = 10000^(-2*lane/64) = exp2(-lane/32 * log2(1e4))
  float fr = (float)s * exp2f(-(float)lane * (13.2877123795f / 32.0f));
  float sn, cs;
  __sincosf(fr, &sn, &cs);
  float y0 = x0 * cs - x1 * sn;   // d < 32 : x*cos - x[d+32]*sin
  float y1 = x1 * cs + x0 * sn;   // d >= 32: x*cos + x[d-32]*sin
  float ss = y0 * y0 + y1 * y1;
#pragma unroll
  for (int m = 1; m < 32; m <<= 1) ss += __shfl_xor(ss, m, 32);
  float inv = 1.0f / fmaxf(sqrtf(ss), 1e-6f);
  float e0 = s_qk[h * HD + lane] * SQRT_DIM;
  float e1 = s_qk[h * HD + lane + 32] * SQRT_DIM;
  bf16* op = Out + ((size_t)(b * HEADS + h) * SEQ + s) * HD;
  op[lane]      = (bf16)(y0 * inv * e0);
  op[lane + 32] = (bf16)(y1 * inv * e1);
}

// ---------------------------------------------------------------------------
// Causal flash attention, WMMA for QK^T and PV.
// Grid: (SEQ/128, HEADS, BATCH), 256 threads; wave handles 16 query rows.
// Qh/Kh bf16 [B,H,S,HD], Vt bf16 [B,H,HD,S], O f32 [B,S,H*HD].
// ---------------------------------------------------------------------------
__global__ __launch_bounds__(256) void flash_attn(
    const bf16* __restrict__ Qh, const bf16* __restrict__ Kh,
    const bf16* __restrict__ Vt, float* __restrict__ O) {
  __shared__ __attribute__((aligned(16))) bf16 Pl[8][16 * 40]; // per-wave P tile

  const int tid  = threadIdx.x;
  const int wave = tid >> 5;
  const int lane = tid & 31;
  const int half = lane >> 4;
  const int lr   = lane & 15;
  const int h = blockIdx.y;
  const int b = blockIdx.z;
  const int qb = blockIdx.x * 128 + wave * 16;

  const bf16* qbase = Qh + ((size_t)(b * HEADS + h) * SEQ + qb) * HD;
  const bf16* kbase = Kh + (size_t)(b * HEADS + h) * SEQ * HD;
  const bf16* vbase = Vt + (size_t)(b * HEADS + h) * HD * SEQ;
  bf16* pl = &Pl[wave][0];

  // Q A-fragments (16x64 = two 16x32 frags)
  v16bf qa[2];
#pragma unroll
  for (int t = 0; t < 2; ++t) {
    const bf16* p = qbase + (size_t)lr * HD + t * 32 + half * 8;
    qa[t] = load_frag(p, p + 16);
  }

  float mrow[8], lrow[8];
  v8f acc[4];
#pragma unroll
  for (int r = 0; r < 8; ++r) { mrow[r] = -3.0e38f; lrow[r] = 0.0f; }
#pragma unroll
  for (int nt = 0; nt < 4; ++nt) acc[nt] = zero8();

  const int nchunks = (qb + 16 + 31) >> 5;   // 32-key chunks touching causal band
  for (int c = 0; c < nchunks; ++c) {
    const int kstart = c * 32;
    // S = Q K^T  (two 16-key column tiles, K-dim 64 = 2 WMMA each)
    v8f st[2];
#pragma unroll
    for (int nh = 0; nh < 2; ++nh) {
      v8f s = zero8();
#pragma unroll
      for (int t = 0; t < 2; ++t) {
        const bf16* p =
            kbase + (size_t)(kstart + nh * 16 + lr) * HD + t * 32 + half * 8;
        v16bf kf = load_frag(p, p + 16);
        s = wmma_bf16(qa[t], kf, s);
      }
      st[nh] = s;
    }
    // scale + causal mask (C layout: lane = key col, VGPR r = query row)
#pragma unroll
    for (int r = 0; r < 8; ++r) {
      int qrow = qb + half * 8 + r;
#pragma unroll
      for (int nh = 0; nh < 2; ++nh) {
        int key = kstart + nh * 16 + lr;
        float v = st[nh][r] * SQRT_HD;
        st[nh][r] = (key <= qrow) ? v : -3.0e38f;
      }
    }
    // online softmax update (row stats replicated across 16-lane halves)
    float sc[8];
#pragma unroll
    for (int r = 0; r < 8; ++r) {
      float tm = fmaxf(st[0][r], st[1][r]);
#pragma unroll
      for (int m = 1; m < 16; m <<= 1) tm = fmaxf(tm, __shfl_xor(tm, m, 32));
      float mn = fmaxf(mrow[r], tm);
      sc[r] = __expf(mrow[r] - mn);
      mrow[r] = mn;
      float p0 = __expf(st[0][r] - mn);
      float p1 = __expf(st[1][r] - mn);
      st[0][r] = p0; st[1][r] = p1;
      float rs = p0 + p1;
#pragma unroll
      for (int m = 1; m < 16; m <<= 1) rs += __shfl_xor(rs, m, 32);
      lrow[r] = lrow[r] * sc[r] + rs;
    }
#pragma unroll
    for (int nt = 0; nt < 4; ++nt)
#pragma unroll
      for (int r = 0; r < 8; ++r) acc[nt][r] *= sc[r];

    // P: C layout -> A layout via wave-private LDS round-trip
#pragma unroll
    for (int nh = 0; nh < 2; ++nh)
#pragma unroll
      for (int r = 0; r < 8; ++r)
        pl[(half * 8 + r) * 40 + nh * 16 + lr] = (bf16)st[nh][r];
    asm volatile("s_wait_dscnt 0" ::: "memory");
    v16bf pa;
    {
      const bf16* p = &pl[lr * 40 + half * 8];
      pa = load_frag(p, p + 16);
    }
    // O += P (16x32) * V (32x64): 4 N-tiles of head dims
#pragma unroll
    for (int nt = 0; nt < 4; ++nt) {
      const bf16* p = vbase + (size_t)(nt * 16 + lr) * SEQ + kstart + half * 8;
      v16bf vf = load_frag(p, p + 16);
      acc[nt] = wmma_bf16(pa, vf, acc[nt]);
    }
  }

  // epilogue: divide by softmax denominator, scatter to [B,S,H*HD]
#pragma unroll
  for (int nt = 0; nt < 4; ++nt)
#pragma unroll
    for (int r = 0; r < 8; ++r) {
      int qrow = qb + half * 8 + r;
      int d = nt * 16 + lr;
      O[((size_t)b * SEQ + qrow) * DIM + h * HD + d] =
          acc[nt][r] / fmaxf(lrow[r], 1e-20f);
    }
}

// ---------------------------------------------------------------------------
// h_out = cosine_norm(h + alpha * (cosine_norm(y) - h)), one block per token.
// ---------------------------------------------------------------------------
__device__ __forceinline__ float block_sum(float v, float* red) {
#pragma unroll
  for (int m = 1; m < 32; m <<= 1) v += __shfl_xor(v, m, 32);
  int wave = threadIdx.x >> 5, lane = threadIdx.x & 31;
  if (lane == 0) red[wave] = v;
  __syncthreads();
  float r = (threadIdx.x < 8) ? red[threadIdx.x] : 0.0f;
  if (wave == 0) {
#pragma unroll
    for (int m = 1; m < 8; m <<= 1) r += __shfl_xor(r, m, 32);
    if (lane == 0) red[0] = r;
  }
  __syncthreads();
  v = red[0];
  __syncthreads();
  return v;
}

__global__ __launch_bounds__(256) void residual_cosnorm(
    const float* __restrict__ Hin, const float* __restrict__ Y,
    const float* __restrict__ alpha, float* __restrict__ Hout,
    bf16* __restrict__ HoutBf) {
  __shared__ float red[8];
  size_t t = blockIdx.x;
  const float* y  = Y + t * DIM;
  const float* hh = Hin + t * DIM;
  float yv[4], hv[4], ss = 0.0f;
#pragma unroll
  for (int i = 0; i < 4; ++i) {
    int c = threadIdx.x + i * 256;
    yv[i] = y[c]; hv[i] = hh[c];
    ss += yv[i] * yv[i];
  }
  ss = block_sum(ss, red);
  float inv1 = 1.0f / fmaxf(sqrtf(ss), 1e-6f);
  float z[4]; float ss2 = 0.0f;
#pragma unroll
  for (int i = 0; i < 4; ++i) {
    int c = threadIdx.x + i * 256;
    z[i] = hv[i] + alpha[c] * (yv[i] * inv1 - hv[i]);
    ss2 += z[i] * z[i];
  }
  ss2 = block_sum(ss2, red);
  float inv2 = 1.0f / fmaxf(sqrtf(ss2), 1e-6f);
#pragma unroll
  for (int i = 0; i < 4; ++i) {
    int c = threadIdx.x + i * 256;
    float o = z[i] * inv2;
    Hout[t * DIM + c] = o;
    if (HoutBf) HoutBf[t * DIM + c] = (bf16)o;
  }
}

// g = (u * s_u) * silu(v * s_v * sqrt(DIM)) -> bf16 (padded cols -> 0)
__global__ void mlp_gate(const float* __restrict__ U, const float* __restrict__ Vg,
                         const float* __restrict__ su, const float* __restrict__ sv,
                         bf16* __restrict__ G) {
  size_t i = (size_t)blockIdx.x * 256 + threadIdx.x;
  size_t total = (size_t)NTOK * HIDP;
  if (i >= total) return;
  int c = (int)(i % HIDP);
  float g = 0.0f;
  if (c < HID) {
    float u  = U[i] * su[c];
    float vv = Vg[i] * sv[c] * SQRT_DIM;
    g = u * (vv / (1.0f + __expf(-vv)));
  }
  G[i] = (bf16)g;
}

// ---------------------------------------------------------------------------
// Launcher
// ---------------------------------------------------------------------------
static inline size_t rup(size_t x) { return (x + 255) & ~(size_t)255; }

extern "C" void kernel_launch(void* const* d_in, const int* in_sizes, int n_in,
                              void* d_out, int out_size, void* d_ws, size_t ws_size,
                              hipStream_t stream) {
  const float* x     = (const float*)d_in[0];
  const float* Wq    = (const float*)d_in[1];
  const float* Wk    = (const float*)d_in[2];
  const float* Wv    = (const float*)d_in[3];
  const float* Wo    = (const float*)d_in[4];
  const float* s_qk  = (const float*)d_in[5];
  const float* a_A   = (const float*)d_in[6];
  const float* Wup   = (const float*)d_in[7];
  const float* Wgate = (const float*)d_in[8];
  const float* Wdown = (const float*)d_in[9];
  const float* s_u   = (const float*)d_in[10];
  const float* s_v   = (const float*)d_in[11];
  const float* a_M   = (const float*)d_in[12];
  float* out = (float*)d_out;

  char* w = (char*)d_ws;
  auto alloc = [&](size_t bytes) -> char* { char* p = w; w += rup(bytes); return p; };

  bf16* xb     = (bf16*)alloc((size_t)NTOK * DIM * 2);
  bf16* WqT    = (bf16*)alloc((size_t)DIM * DIM * 2);
  bf16* WkT    = (bf16*)alloc((size_t)DIM * DIM * 2);
  bf16* WvT    = (bf16*)alloc((size_t)DIM * DIM * 2);
  bf16* WoT    = (bf16*)alloc((size_t)DIM * DIM * 2);
  bf16* WupT   = (bf16*)alloc((size_t)HIDP * DIM * 2);
  bf16* WgateT = (bf16*)alloc((size_t)HIDP * DIM * 2);
  bf16* WdownT = (bf16*)alloc((size_t)DIM * HIDP * 2);
  float* qf    = (float*)alloc((size_t)NTOK * DIM * 4);
  float* kf    = (float*)alloc((size_t)NTOK * DIM * 4);
  float* vf    = (float*)alloc((size_t)NTOK * DIM * 4);
  bf16* qbf    = (bf16*)alloc((size_t)NTOK * DIM * 2);
  bf16* kbf    = (bf16*)alloc((size_t)NTOK * DIM * 2);
  bf16* vtb    = (bf16*)alloc((size_t)NTOK * DIM * 2);
  bf16* aob    = (bf16*)alloc((size_t)NTOK * DIM * 2);
  bf16* h1b    = (bf16*)alloc((size_t)NTOK * DIM * 2);
  float* uf    = (float*)alloc((size_t)NTOK * HIDP * 4);
  float* vgf   = (float*)alloc((size_t)NTOK * HIDP * 4);
  bf16* gb     = (bf16*)alloc((size_t)NTOK * HIDP * 2);
  // buffer reuse after producers are consumed:
  float* ao    = qf;   // flash-attention output
  float* aproj = kf;   // attention projection
  float* h1    = vf;   // post-attention hidden state
  float* mout  = uf;   // MLP down-proj output

  const int T = 256;
  auto blocks1d = [](size_t n) { return (unsigned)((n + 255) / 256); };

  // 1) precision conversion + weight transposes
  cvt_f32_bf16<<<blocks1d((size_t)NTOK * DIM), T, 0, stream>>>(x, xb, (size_t)NTOK * DIM);
  transpose_pad_bf16<<<blocks1d((size_t)DIM * DIM), T, 0, stream>>>(Wq, WqT, DIM, DIM, DIM, DIM);
  transpose_pad_bf16<<<blocks1d((size_t)DIM * DIM), T, 0, stream>>>(Wk, WkT, DIM, DIM, DIM, DIM);
  transpose_pad_bf16<<<blocks1d((size_t)DIM * DIM), T, 0, stream>>>(Wv, WvT, DIM, DIM, DIM, DIM);
  transpose_pad_bf16<<<blocks1d((size_t)DIM * DIM), T, 0, stream>>>(Wo, WoT, DIM, DIM, DIM, DIM);
  transpose_pad_bf16<<<blocks1d((size_t)HIDP * DIM), T, 0, stream>>>(Wup, WupT, DIM, HID, DIM, HIDP);
  transpose_pad_bf16<<<blocks1d((size_t)HIDP * DIM), T, 0, stream>>>(Wgate, WgateT, DIM, HID, DIM, HIDP);
  transpose_pad_bf16<<<blocks1d((size_t)DIM * HIDP), T, 0, stream>>>(Wdown, WdownT, HID, DIM, HIDP, DIM);

  // 2) QKV projections
  dim3 gD(DIM / GBN, NTOK / GBM);
  gemm_bf16_wmma<<<gD, T, 0, stream>>>(xb, WqT, qf, NTOK, DIM, DIM);
  gemm_bf16_wmma<<<gD, T, 0, stream>>>(xb, WkT, kf, NTOK, DIM, DIM);
  gemm_bf16_wmma<<<gD, T, 0, stream>>>(xb, WvT, vf, NTOK, DIM, DIM);

  // 3) RoPE + cosine norm on q,k; transpose v
  unsigned ropeBlocks = (unsigned)(((size_t)NTOK * HEADS + 7) / 8);
  rope_cosnorm<<<ropeBlocks, T, 0, stream>>>(qf, s_qk, qbf);
  rope_cosnorm<<<ropeBlocks, T, 0, stream>>>(kf, s_qk, kbf);
  v_transpose_bf16<<<blocks1d((size_t)NTOK * DIM), T, 0, stream>>>(vf, vtb);

  // 4) causal flash attention
  dim3 gFA(SEQ / 128, HEADS, BATCH);
  flash_attn<<<gFA, T, 0, stream>>>(qbf, kbf, vtb, ao);

  // 5) output projection + residual cosine norm
  cvt_f32_bf16<<<blocks1d((size_t)NTOK * DIM), T, 0, stream>>>(ao, aob, (size_t)NTOK * DIM);
  gemm_bf16_wmma<<<gD, T, 0, stream>>>(aob, WoT, aproj, NTOK, DIM, DIM);
  residual_cosnorm<<<NTOK, T, 0, stream>>>(x, aproj, a_A, h1, h1b);

  // 6) MLP
  dim3 gUp(HIDP / GBN, NTOK / GBM);
  gemm_bf16_wmma<<<gUp, T, 0, stream>>>(h1b, WupT, uf, NTOK, HIDP, DIM);
  gemm_bf16_wmma<<<gUp, T, 0, stream>>>(h1b, WgateT, vgf, NTOK, HIDP, DIM);
  mlp_gate<<<blocks1d((size_t)NTOK * HIDP), T, 0, stream>>>(uf, vgf, s_u, s_v, gb);
  gemm_bf16_wmma<<<gD, T, 0, stream>>>(gb, WdownT, mout, NTOK, DIM, HIDP);

  // 7) final residual cosine norm -> output
  residual_cosnorm<<<NTOK, T, 0, stream>>>(h1, mout, a_M, out, (bf16*)nullptr);

  (void)in_sizes; (void)n_in; (void)out_size; (void)ws_size;
}
